// CrossAttention_49821620633894
// MI455X (gfx1250) — matrix operand order repaired and checked by reading, not compiled, not run on hardware
//
#include <hip/hip_runtime.h>
#include <hip/hip_bf16.h>
#include <stdint.h>

typedef _Float16 f16;
typedef __attribute__((ext_vector_type(8)))  _Float16 v8h;
typedef __attribute__((ext_vector_type(16))) _Float16 v16h;
typedef __attribute__((ext_vector_type(8)))  float    v8f;
typedef __attribute__((ext_vector_type(4)))  int      v4i;

#define B_    16
#define H_    12
#define F_    16
#define NP    196
#define NQ    32
#define DIM   768
#define DHEAD 64
#define NKV   3136        // F_ * NP
#define FPAD  224         // frame length padded to a multiple of 32
#define QPAD  48          // padded queries per (b,h)
#define NOUT  497         // 1 + F_*(NQ-1)
#define MPADO 8064        // 7952 output rows padded to multiple of 128

// ---------------- gfx1250 async-to-LDS path (guarded) ----------------

#if __has_builtin(__builtin_amdgcn_global_load_async_to_lds_b128) && \
    __has_builtin(__builtin_amdgcn_s_wait_asynccnt)
#define HAVE_ASYNC_LDS 1
#else
#define HAVE_ASYNC_LDS 0
#endif

__device__ __forceinline__ void copy16_to_lds(f16* lds_dst, const f16* gsrc) {
#if HAVE_ASYNC_LDS
  __builtin_amdgcn_global_load_async_to_lds_b128((v4i*)gsrc, (v4i*)lds_dst, 0, 0);
#else
  *(uint4*)lds_dst = *(const uint4*)gsrc;
#endif
}

__device__ __forceinline__ void async_copy_wait() {
#if HAVE_ASYNC_LDS
  __builtin_amdgcn_s_wait_asynccnt(0);
#endif
}

// ---------------- WMMA fragment helpers (layouts per ISA 7.12.2) ----------------

__device__ __forceinline__ v16h load_a_frag(const f16* A, int lda, int m0, int k0) {
  const int lane = threadIdx.x & 31;
  const int m    = m0 + (lane & 15);
  const int hb   = (lane >> 4) << 3;            // 0 or 8
  const f16* p = A + (size_t)m * lda + k0 + hb;
  union { v16h v; v8h h[2]; } u;
  u.h[0] = *(const v8h*)(p);                    // K = k0+hb .. +7
  u.h[1] = *(const v8h*)(p + 16);               // K = k0+16+hb .. +7
  return u.v;
}

// B stored transposed: Bt[n][k], n = output column
__device__ __forceinline__ v16h load_b_frag(const f16* Bt, int ldb, int n0, int k0) {
  const int lane = threadIdx.x & 31;
  const int n    = n0 + (lane & 15);
  const int hb   = (lane >> 4) << 3;            // 0 or 8
  const f16* p = Bt + (size_t)n * ldb + k0 + hb;
  union { v16h v; v8h h[2]; } u;
  u.h[0] = *(const v8h*)(p);
  u.h[1] = *(const v8h*)(p + 16);
  return u.v;
}

__device__ __forceinline__ v8f wmma16(v16h a, v16h b, v8f c) {
  return __builtin_amdgcn_wmma_f32_16x16x32_f16(false, a, false, b, (short)0, c, false, false);
}

// One wave accumulates a 64x64 tile (4x4 fragments, 16 WMMA per 32-deep k-step)
__device__ __forceinline__ void gemm_acc_64x64(const f16* A, int lda, const f16* Bt, int ldb,
                                               int m0, int n0, int K, v8f acc[16]) {
  for (int k = 0; k < K; k += 32) {
    v16h a[4], b[4];
#pragma unroll
    for (int i = 0; i < 4; ++i) a[i] = load_a_frag(A, lda, m0 + 16 * i, k);
#pragma unroll
    for (int i = 0; i < 4; ++i) b[i] = load_b_frag(Bt, ldb, n0 + 16 * i, k);
#pragma unroll
    for (int mi = 0; mi < 4; ++mi)
#pragma unroll
      for (int ni = 0; ni < 4; ++ni)
        acc[mi * 4 + ni] = wmma16(a[mi], b[ni], acc[mi * 4 + ni]);
  }
}

// 8 waves / block: 2 (M) x 4 (N) wave grid -> block tile 128 x 256
__device__ __forceinline__ void wave_tile64(int& m0, int& n0) {
  const int w = threadIdx.x >> 5;
  m0 = blockIdx.y * 128 + (w >> 2) * 64;
  n0 = blockIdx.x * 256 + (w & 3)  * 64;
}

// ---------------- utility kernels ----------------

__global__ void k_zero(uint32_t* p, size_t n) {
  for (size_t i = (size_t)blockIdx.x * 256 + threadIdx.x; i < n; i += (size_t)gridDim.x * 256)
    p[i] = 0u;
}

__global__ void k_cvt(const float* __restrict__ src, f16* __restrict__ dst, size_t n) {
  for (size_t i = (size_t)blockIdx.x * 256 + threadIdx.x; i < n; i += (size_t)gridDim.x * 256)
    dst[i] = (f16)src[i];
}

// src: K x N row-major (fp32) -> dst: N x K row-major (f16)
__global__ void k_cvt_T(const float* __restrict__ src, f16* __restrict__ dst, int K, int N) {
  size_t total = (size_t)K * N;
  for (size_t i = (size_t)blockIdx.x * 256 + threadIdx.x; i < total; i += (size_t)gridDim.x * 256) {
    int k = (int)(i / N), n = (int)(i % N);
    dst[(size_t)n * K + k] = (f16)src[i];
  }
}

// ---------------- projection GEMMs ----------------

// Q = scale * (question @ Wq); scatter into Qh[b][h][q][d] (q padded to QPAD)
__global__ __launch_bounds__(256) void k_qproj(const f16* __restrict__ qf,
                                               const f16* __restrict__ WqT,
                                               f16* __restrict__ Qh) {
  int m0, n0; wave_tile64(m0, n0);
  v8f acc[16] = {};
  gemm_acc_64x64(qf, DIM, WqT, DIM, m0, n0, DIM, acc);
  const int lane = threadIdx.x & 31, half = lane >> 4;
  const float scale = 0.125f;     // 64^-0.5
  for (int mi = 0; mi < 4; ++mi)
    for (int ni = 0; ni < 4; ++ni)
      for (int v = 0; v < 8; ++v) {
        int m = m0 + mi * 16 + half * 8 + v;
        int n = n0 + ni * 16 + (lane & 15);
        int b = m >> 5, q = m & 31;
        int h = n >> 6, d = n & 63;
        Qh[(((size_t)b * H_ + h) * QPAD + q) * DHEAD + d] = (f16)(acc[mi * 4 + ni][v] * scale);
      }
}

// [K | V] = x @ Wkv; K -> Kh[b][h][f][j(FPAD)][d],  V -> Vt[b][h][d][f*FPAD + j]
__global__ __launch_bounds__(256) void k_kvproj(const f16* __restrict__ xf,
                                                const f16* __restrict__ WkvT,
                                                f16* __restrict__ Kh,
                                                f16* __restrict__ Vt) {
  int m0, n0; wave_tile64(m0, n0);
  v8f acc[16] = {};
  gemm_acc_64x64(xf, DIM, WkvT, DIM, m0, n0, DIM, acc);
  const int lane = threadIdx.x & 31, half = lane >> 4;
  for (int mi = 0; mi < 4; ++mi)
    for (int ni = 0; ni < 4; ++ni)
      for (int v = 0; v < 8; ++v) {
        int m = m0 + mi * 16 + half * 8 + v;
        int n = n0 + ni * 16 + (lane & 15);
        int b = m / NKV, j = m % NKV;
        int fr = j / NP, jj = j % NP;
        float val = acc[mi * 4 + ni][v];
        if (n < DIM) {
          int h = n >> 6, d = n & 63;
          size_t bh = (size_t)b * H_ + h;
          Kh[((bh * F_ + fr) * FPAD + jj) * DHEAD + d] = (f16)val;
        } else {
          int nv = n - DIM, h = nv >> 6, d = nv & 63;
          size_t bh = (size_t)b * H_ + h;
          Vt[(bh * DHEAD + d) * (size_t)(F_ * FPAD) + (size_t)fr * FPAD + jj] = (f16)val;
        }
      }
}

// out = O @ Wproj + bproj  (fp32 output)
__global__ __launch_bounds__(256) void k_outproj(const f16* __restrict__ O,
                                                 const f16* __restrict__ WpT,
                                                 const float* __restrict__ bproj,
                                                 float* __restrict__ out) {
  int m0, n0; wave_tile64(m0, n0);
  v8f acc[16] = {};
  gemm_acc_64x64(O, DIM, WpT, DIM, m0, n0, DIM, acc);
  const int lane = threadIdx.x & 31, half = lane >> 4;
  for (int mi = 0; mi < 4; ++mi)
    for (int ni = 0; ni < 4; ++ni)
      for (int v = 0; v < 8; ++v) {
        int m = m0 + mi * 16 + half * 8 + v;
        int n = n0 + ni * 16 + (lane & 15);
        if (m < B_ * NOUT)
          out[(size_t)m * DIM + n] = acc[mi * 4 + ni][v] + bproj[n];
      }
}

// ---------------- fused per-frame attention ----------------

__global__ __launch_bounds__(256) void k_attention(const f16* __restrict__ Qh,
                                                   const f16* __restrict__ Kh,
                                                   const f16* __restrict__ Vt,
                                                   f16* __restrict__ O) {
  const int fr = blockIdx.x % F_;
  const int h  = (blockIdx.x / F_) % H_;
  const int b  = blockIdx.x / (F_ * H_);
  const size_t bh = (size_t)b * H_ + h;

  __shared__ __attribute__((aligned(16))) float S[32 * FPAD];
  __shared__ __attribute__((aligned(16))) f16   P[32 * FPAD];
  __shared__ __attribute__((aligned(16))) f16   KVsh[FPAD * DHEAD];  // K tile, then V tile
  __shared__ float red[32][8];
  __shared__ float smax[32];
  __shared__ float sinv[32];

  const int tid = threadIdx.x, w = tid >> 5, lane = tid & 31, half = lane >> 4;
  const f16* Qbase = Qh + (bh * QPAD + 1) * DHEAD;               // queries 1..31 (+ zero pad)
  const f16* Kbase = Kh + ((bh * F_ + fr) * (size_t)FPAD) * DHEAD;
  const f16* Vbase = Vt + (bh * DHEAD) * (size_t)(F_ * FPAD) + (size_t)fr * FPAD;

  // stage K tile (FPAD x DHEAD, contiguous 28KB) into LDS via async loads
  for (int c = tid; c < (FPAD * DHEAD) / 8; c += 256)
    copy16_to_lds(KVsh + c * 8, Kbase + c * 8);
  async_copy_wait();
  __syncthreads();

  // S = Q_ @ K^T : 2 x 13 tiles of 16x16, K-depth 64 (K from LDS)
  for (int t = w; t < 26; t += 8) {
    int mt = t & 1, nt = t >> 1;
    v8f acc = {};
    for (int k = 0; k < DHEAD; k += 32) {
      v16h a  = load_a_frag(Qbase, DHEAD, mt * 16, k);
      v16h bf = load_b_frag(KVsh, DHEAD, nt * 16, k);
      acc = wmma16(a, bf, acc);
    }
    int col = nt * 16 + (lane & 15);
    for (int v = 0; v < 8; ++v)
      S[(mt * 16 + half * 8 + v) * FPAD + col] = acc[v];
  }
  __syncthreads();   // all waves done with K tile; S complete

  // stage V tile (DHEAD x FPAD rows, global row stride F_*FPAD) over the K buffer;
  // softmax below hides the async copy latency
  for (int c = tid; c < (DHEAD * FPAD) / 8; c += 256) {
    int d = c / (FPAD / 8), cc = c % (FPAD / 8);
    copy16_to_lds(KVsh + d * FPAD + cc * 8, Vbase + (size_t)d * (F_ * FPAD) + cc * 8);
  }

  // mask padded key columns
  for (int idx = tid; idx < 32 * (FPAD - NP); idx += 256) {
    int row = idx / (FPAD - NP);
    int col = NP + idx % (FPAD - NP);
    S[row * FPAD + col] = -3.0e38f;
  }
  __syncthreads();

  // row softmax (8 threads per row, 28 cols each)
  const int row = tid >> 3, sub = tid & 7, c0 = sub * (FPAD / 8);
  float lmax = -3.0e38f;
  for (int c = 0; c < FPAD / 8; ++c) lmax = fmaxf(lmax, S[row * FPAD + c0 + c]);
  red[row][sub] = lmax;
  __syncthreads();
  if (sub == 0) {
    float m = red[row][0];
    for (int s = 1; s < 8; ++s) m = fmaxf(m, red[row][s]);
    smax[row] = m;
  }
  __syncthreads();
  const float rmax = smax[row];
  float lsum = 0.f;
  for (int c = 0; c < FPAD / 8; ++c) {
    float e = __expf(S[row * FPAD + c0 + c] - rmax);
    P[row * FPAD + c0 + c] = (f16)e;
    lsum += e;
  }
  red[row][sub] = lsum;
  __syncthreads();
  if (sub == 0) {
    float s = 0.f;
    for (int q = 0; q < 8; ++q) s += red[row][q];
    sinv[row] = 1.0f / s;
  }
  async_copy_wait();  // V tile resident
  __syncthreads();

  // Otile = P @ V : 2 x 4 tiles of 16x16, K-depth FPAD (V from LDS, padded cols zero)
  {
    int mt = w & 1, nt = w >> 1;
    v8f acc = {};
    for (int k = 0; k < FPAD; k += 32) {
      v16h a  = load_a_frag(P, FPAD, mt * 16, k);
      v16h bf = load_b_frag(KVsh, FPAD, nt * 16, k);
      acc = wmma16(a, bf, acc);
    }
    int d = nt * 16 + (lane & 15);
    for (int v = 0; v < 8; ++v) {
      int i = mt * 16 + half * 8 + v;
      if (i < NQ - 1) {
        size_t orow = (size_t)b * NOUT + 1 + (size_t)fr * (NQ - 1) + i;
        O[orow * DIM + h * DHEAD + d] = (f16)(acc[v] * sinv[i]);
      }
    }
  }
}

// ---------------- CLS-token attention (1 query vs 3136 keys, VALU) ----------------

__global__ __launch_bounds__(256) void k_cls(const f16* __restrict__ Qh,
                                             const f16* __restrict__ Kh,
                                             const f16* __restrict__ Vt,
                                             f16* __restrict__ O) {
  const int h = blockIdx.x % H_;
  const int b = blockIdx.x / H_;
  const size_t bh = (size_t)b * H_ + h;
  __shared__ float p[NKV];
  __shared__ float qs[DHEAD];
  __shared__ float red[256];
  const int tid = threadIdx.x;
  if (tid < DHEAD) qs[tid] = (float)Qh[(bh * QPAD) * DHEAD + tid];
  __syncthreads();

  float lmax = -3.0e38f;
  for (int j = tid; j < NKV; j += 256) {
    int fr = j / NP, jj = j % NP;
    const f16* krow = Kh + ((bh * F_ + fr) * (size_t)FPAD + jj) * DHEAD;
    float s = 0.f;
    for (int d = 0; d < DHEAD; ++d) s += qs[d] * (float)krow[d];
    p[j] = s;
    lmax = fmaxf(lmax, s);
  }
  red[tid] = lmax; __syncthreads();
  for (int st = 128; st > 0; st >>= 1) { if (tid < st) red[tid] = fmaxf(red[tid], red[tid + st]); __syncthreads(); }
  const float m = red[0]; __syncthreads();

  float lsum = 0.f;
  for (int j = tid; j < NKV; j += 256) { float e = __expf(p[j] - m); p[j] = e; lsum += e; }
  red[tid] = lsum; __syncthreads();
  for (int st = 128; st > 0; st >>= 1) { if (tid < st) red[tid] += red[tid + st]; __syncthreads(); }
  const float inv = 1.0f / red[0]; __syncthreads();

  const int d = tid & 63, qtr = tid >> 6;
  const f16* vrow = Vt + (bh * DHEAD + d) * (size_t)(F_ * FPAD);
  float acc = 0.f;
  for (int j = qtr * (NKV / 4); j < (qtr + 1) * (NKV / 4); ++j) {
    int fr = j / NP, jj = j % NP;
    acc += p[j] * (float)vrow[(size_t)fr * FPAD + jj];
  }
  red[tid] = acc; __syncthreads();
  if (qtr == 0) {
    float tot = red[d] + red[64 + d] + red[128 + d] + red[192 + d];
    O[((size_t)b * NOUT) * DIM + h * DHEAD + d] = (f16)(tot * inv);
  }
}

// ---------------- host launch ----------------

extern "C" void kernel_launch(void* const* d_in, const int* in_sizes, int n_in,
                              void* d_out, int out_size, void* d_ws, size_t ws_size,
                              hipStream_t stream) {
  (void)in_sizes; (void)n_in; (void)out_size; (void)ws_size;
  const float* x        = (const float*)d_in[0];
  const float* question = (const float*)d_in[1];
  const float* Wq       = (const float*)d_in[2];
  const float* Wkv      = (const float*)d_in[3];
  const float* Wproj    = (const float*)d_in[4];
  const float* bproj    = (const float*)d_in[5];
  float* out = (float*)d_out;

  char* ws = (char*)d_ws;
  size_t off = 0;
  auto take = [&](size_t bytes) -> char* {
    char* p = ws + off;
    off += (bytes + 255) & ~(size_t)255;
    return p;
  };

  const size_t N_X   = (size_t)B_ * NKV * DIM;              // 38,535,168
  const size_t N_Q   = (size_t)B_ * NQ * DIM;               // 393,216
  const size_t N_QH  = (size_t)B_ * H_ * QPAD * DHEAD;      // 589,824
  const size_t N_KH  = (size_t)B_ * H_ * F_ * FPAD * DHEAD; // 44,040,192
  const size_t N_VT  = N_KH;
  const size_t N_O   = (size_t)MPADO * DIM;                 // 6,193,152

  f16* xf   = (f16*)take(N_X  * 2);
  f16* qf   = (f16*)take(N_Q  * 2);
  f16* WqT  = (f16*)take((size_t)DIM * DIM * 2);
  f16* WkvT = (f16*)take((size_t)DIM * 2 * DIM * 2);
  f16* WpT  = (f16*)take((size_t)DIM * DIM * 2);
  f16* Qh   = (f16*)take(N_QH * 2);
  f16* Kh   = (f16*)take(N_KH * 2);
  f16* Vt   = (f16*)take(N_VT * 2);
  f16* O    = (f16*)take(N_O  * 2);

  dim3 blk(256);
  // zero buffers whose padding participates in WMMA (NaN safety)
  k_zero<<<2048, blk, 0, stream>>>((uint32_t*)Vt, N_VT / 2);
  k_zero<<<256,  blk, 0, stream>>>((uint32_t*)Qh, N_QH / 2);
  k_zero<<<1024, blk, 0, stream>>>((uint32_t*)O,  N_O  / 2);

  // fp32 -> f16 conversions / weight transposes
  k_cvt  <<<4096, blk, 0, stream>>>(x, xf, N_X);
  k_cvt  <<<512,  blk, 0, stream>>>(question, qf, N_Q);
  k_cvt_T<<<1024, blk, 0, stream>>>(Wq,    WqT,  DIM, DIM);
  k_cvt_T<<<2048, blk, 0, stream>>>(Wkv,   WkvT, DIM, 2 * DIM);
  k_cvt_T<<<1024, blk, 0, stream>>>(Wproj, WpT,  DIM, DIM);

  // projections (WMMA, 64x64 wave tiles)
  k_qproj <<<dim3(DIM / 256, (B_ * NQ) / 128),        blk, 0, stream>>>(qf, WqT, Qh);
  k_kvproj<<<dim3((2 * DIM) / 256, (B_ * NKV) / 128), blk, 0, stream>>>(xf, WkvT, Kh, Vt);

  // attention (WMMA + async-to-LDS staging) + cls
  k_attention<<<B_ * H_ * F_, blk, 0, stream>>>(Qh, Kh, Vt, O);
  k_cls      <<<B_ * H_,      blk, 0, stream>>>(Qh, Kh, Vt, O);

  // output projection (WMMA) + bias
  k_outproj<<<dim3(DIM / 256, MPADO / 128), blk, 0, stream>>>(O, WpT, bproj, out);
}
